// HyperND_2241972928550
// MI455X (gfx1250) — matrix operand order repaired and checked by reading, not compiled, not run on hardware
//
#include <hip/hip_runtime.h>
#include <hip/hip_bf16.h>

typedef __bf16 v8bf  __attribute__((ext_vector_type(8)));
typedef __bf16 v16bf __attribute__((ext_vector_type(16)));
typedef float  v8f   __attribute__((ext_vector_type(8)));

#define NN   50000
#define MM   10000
#define DF   128
#define HIDN 256
#define OUTN 40

union V16U { v16bf v; struct { v8bf lo, hi; } h; };

// Load one 16-bit A-fragment half-row: two contiguous 8x bf16 runs (K..K+7, K+16..K+23)
__device__ __forceinline__ v16bf load_frag16(const __bf16* p) {
  V16U u;
  u.h.lo = *(const v8bf*)(p);
  u.h.hi = *(const v8bf*)(p + 16);
  return u.v;
}

// ---------------- diffusion kernels ----------------

__global__ void k_deg(const int* __restrict__ V, const int* __restrict__ E,
                      float* __restrict__ D, float* __restrict__ DE, int nnz) {
  int i = blockIdx.x * blockDim.x + threadIdx.x;
  if (i < nnz) {
    unsafeAtomicAdd(&D[V[i]], 1.0f);
    unsafeAtomicAdd(&DE[E[i]], 1.0f);
  }
}

__global__ void k_invdeg(float* __restrict__ D, float* __restrict__ DE) {
  int i = blockIdx.x * blockDim.x + threadIdx.x;
  if (i < NN) D[i] = rsqrtf(D[i]);      // D -> 1/sqrt(deg)
  if (i < MM) DE[i] = 1.0f / DE[i];     // DE -> 1/|e|
}

// Xe[e] += (Z[v] * rsD[v])^2, one wave per incidence, float4 per lane
__global__ void k_v2e(const float* __restrict__ Z, const int* __restrict__ V,
                      const int* __restrict__ E, const float* __restrict__ rsD,
                      float* __restrict__ Xe, int nnz) {
  int lane = threadIdx.x & 31;
  int i = blockIdx.x * (blockDim.x >> 5) + (threadIdx.x >> 5);
  if (i >= nnz) return;
  int v = V[i], e = E[i];
  float rs = rsD[v];
  float4 z = *(const float4*)(Z + (size_t)v * DF + lane * 4);
  float tx = z.x * rs, ty = z.y * rs, tz = z.z * rs, tw = z.w * rs;
  float* dst = Xe + (size_t)e * DF + lane * 4;
  unsafeAtomicAdd(dst + 0, tx * tx);
  unsafeAtomicAdd(dst + 1, ty * ty);
  unsafeAtomicAdd(dst + 2, tz * tz);
  unsafeAtomicAdd(dst + 3, tw * tw);
}

// Xv[v] += Pe[e]
__global__ void k_e2v(const float* __restrict__ Pe, const int* __restrict__ V,
                      const int* __restrict__ E, float* __restrict__ Xv, int nnz) {
  int lane = threadIdx.x & 31;
  int i = blockIdx.x * (blockDim.x >> 5) + (threadIdx.x >> 5);
  if (i >= nnz) return;
  int v = V[i], e = E[i];
  float4 p = *(const float4*)(Pe + (size_t)e * DF + lane * 4);
  float* dst = Xv + (size_t)v * DF + lane * 4;
  unsafeAtomicAdd(dst + 0, p.x);
  unsafeAtomicAdd(dst + 1, p.y);
  unsafeAtomicAdd(dst + 2, p.z);
  unsafeAtomicAdd(dst + 3, p.w);
}

// Pe = sqrt(Xe / |e|) * inv_phi   (== V2E of the normalized node signal)
__global__ void k_pe(const float* __restrict__ Xe, const float* __restrict__ iDE,
                     const float* __restrict__ scal, float* __restrict__ Pe) {
  int i = blockIdx.x * blockDim.x + threadIdx.x;
  if (i < MM * DF) Pe[i] = sqrtf(Xe[i] * iDE[i >> 7]) * scal[1];
}

// G = 0.9 * Xv * rsqrtD + 0.1 * U   (in place on G which holds Xv)
__global__ void k_combine(float* __restrict__ G, const float* __restrict__ rsD,
                          const float* __restrict__ U) {
  int i = blockIdx.x * blockDim.x + threadIdx.x;
  if (i < NN * DF) G[i] = 0.9f * G[i] * rsD[i >> 7] + 0.1f * U[i];
}

// s += sum(Xe / |e|)   -> phi^2/4
__global__ void k_reduce(const float* __restrict__ Xe, const float* __restrict__ iDE,
                         float* __restrict__ s) {
  int i = blockIdx.x * blockDim.x + threadIdx.x;
  float v = 0.0f;
  if (i < MM * DF) v = Xe[i] * iDE[i >> 7];
  for (int o = 16; o > 0; o >>= 1) v += __shfl_down(v, o, 32);
  if ((threadIdx.x & 31) == 0) unsafeAtomicAdd(s, v);
}

__global__ void k_finphi(float* __restrict__ scal) {
  scal[1] = 0.5f * rsqrtf(scal[0]);   // 1/phi = 1/(2*sqrt(s))
}

__global__ void k_initUG(const float* __restrict__ x, const float* __restrict__ scal,
                         float* __restrict__ U, float* __restrict__ G) {
  int i = blockIdx.x * blockDim.x + threadIdx.x;
  if (i < NN * DF) { float xv = x[i]; U[i] = xv * scal[1]; G[i] = xv; }
}

__global__ void k_tobf(const float* __restrict__ G, const float* __restrict__ scal,
                       __bf16* __restrict__ Fbf) {
  int i = blockIdx.x * blockDim.x + threadIdx.x;
  if (i < NN * DF) Fbf[i] = (__bf16)(G[i] * scal[1]);
}

// ---------------- weight pre-swizzle (B-matrix lane layout, ISA 7.12.2) ----------------
// layout: Wsw[(((nb*KB)+kb)*32 + lane)*16 + j] = W[K][N],
//   N = nb*16 + (lane&15), K = kb*32 + (lane>=16 ? 8 : 0) + (j&7) + (j>=8 ? 16 : 0)

__global__ void k_swz_w1(const float* __restrict__ W1, __bf16* __restrict__ W1s) {
  int t = blockIdx.x * blockDim.x + threadIdx.x;
  if (t >= 16 * 4 * 32 * 16) return;
  int j = t & 15, lane = (t >> 4) & 31, kb = (t >> 9) & 3, nb = t >> 11;
  int Ncol = nb * 16 + (lane & 15);
  int K = kb * 32 + ((lane & 16) ? 8 : 0) + (j & 7) + ((j & 8) ? 16 : 0);
  W1s[t] = (__bf16)W1[(size_t)K * HIDN + Ncol];
}

__global__ void k_swz_w2(const float* __restrict__ W2, __bf16* __restrict__ W2s) {
  int t = blockIdx.x * blockDim.x + threadIdx.x;
  if (t >= 4 * 8 * 32 * 16) return;
  int j = t & 15, lane = (t >> 4) & 31, kb = (t >> 9) & 7, nb = t >> 12;
  int Ncol = nb * 16 + (lane & 15);
  int K = kb * 32 + ((lane & 16) ? 8 : 0) + (j & 7) + ((j & 8) ? 16 : 0);
  W2s[t] = (Ncol < OUTN) ? (__bf16)W2[(size_t)K * OUTN + Ncol] : (__bf16)0.0f;
}

// ---------------- fused decoder: GEMM1 + LN + ReLU + GEMM2 (WMMA bf16) ----------------
// 16 rows per workgroup, 4 waves. GEMM1: wave w covers N-tiles 4w..4w+3 (K=128).
// h tile (16x256 f32) lives in LDS; LN over 256; GEMM2: wave w covers N-tile w (K=256).

__global__ void __launch_bounds__(128)
k_decoder(const __bf16* __restrict__ Fbf, const __bf16* __restrict__ W1s,
          const float* __restrict__ b1, const float* __restrict__ gam,
          const float* __restrict__ bet, const __bf16* __restrict__ W2s,
          const float* __restrict__ b2, float* __restrict__ out) {
  __shared__ __align__(32) float  lds_h[16][HIDN];
  __shared__ __align__(32) __bf16 lds_hb[16][HIDN];

  const int lane = threadIdx.x & 31;
  const int wid  = threadIdx.x >> 5;
  const int m15  = lane & 15;
  const int hi   = (lane >> 4) & 1;
  const int row0 = blockIdx.x * 16;

  // A fragments: row (row0 + m15), 4 K-blocks of 32 — shared across this wave's N-tiles
  const __bf16* arow = Fbf + (size_t)(row0 + m15) * DF;
  v16bf afrag[4];
#pragma unroll
  for (int kb = 0; kb < 4; ++kb)
    afrag[kb] = load_frag16(arow + kb * 32 + hi * 8);

#pragma unroll
  for (int nt = 0; nt < 4; ++nt) {
    const int ntile = wid * 4 + nt;
    v8f c = {};
#pragma unroll
    for (int kb = 0; kb < 4; ++kb) {
      v16bf bfrag = *(const v16bf*)(W1s + ((size_t)(ntile * 4 + kb) * 32 + lane) * 16);
      c = __builtin_amdgcn_wmma_f32_16x16x32_bf16(false, afrag[kb], false, bfrag,
                                                  (short)0, c, false, false);
    }
    const int ncol = ntile * 16 + m15;
    const float bb = b1[ncol];
#pragma unroll
    for (int r = 0; r < 8; ++r)
      lds_h[r + hi * 8][ncol] = c[r] + bb;   // C layout: VGPR r -> M = r + 8*hi, N = lane&15
  }
  __syncthreads();

  // LayerNorm + ReLU, one thread per row
  if (threadIdx.x < 16) {
    const int row = threadIdx.x;
    float s = 0.0f, s2 = 0.0f;
    for (int j = 0; j < HIDN; ++j) { float v = lds_h[row][j]; s += v; s2 += v * v; }
    const float mu  = s * (1.0f / HIDN);
    const float var = s2 * (1.0f / HIDN) - mu * mu;
    const float rs  = rsqrtf(var + 1e-5f);
    for (int j = 0; j < HIDN; ++j) {
      float v = (lds_h[row][j] - mu) * rs * gam[j] + bet[j];
      v = v > 0.0f ? v : 0.0f;
      lds_hb[row][j] = (__bf16)v;
    }
  }
  __syncthreads();

  // GEMM2: wave wid -> output N-tile wid (cols wid*16 .. wid*16+15), K = 256
  v8f c2 = {};
  const __bf16* arow2 = &lds_hb[m15][0];
#pragma unroll
  for (int kb = 0; kb < 8; ++kb) {
    v16bf a2 = load_frag16(arow2 + kb * 32 + hi * 8);
    v16bf bf = *(const v16bf*)(W2s + ((size_t)(wid * 8 + kb) * 32 + lane) * 16);
    c2 = __builtin_amdgcn_wmma_f32_16x16x32_bf16(false, a2, false, bf,
                                                 (short)0, c2, false, false);
  }
  const int ncol2 = wid * 16 + m15;
  if (ncol2 < OUTN) {
    const float bb2 = b2[ncol2];
#pragma unroll
    for (int r = 0; r < 8; ++r)
      out[(size_t)(row0 + r + hi * 8) * OUTN + ncol2] = c2[r] + bb2;
  }
}

// ---------------- host orchestration ----------------

extern "C" void kernel_launch(void* const* d_in, const int* in_sizes, int n_in,
                              void* d_out, int out_size, void* d_ws, size_t ws_size,
                              hipStream_t stream) {
  const float* x  = (const float*)d_in[0];
  const int* hei  = (const int*)d_in[1];
  const float* W1 = (const float*)d_in[2];
  const float* b1 = (const float*)d_in[3];
  const float* gm = (const float*)d_in[4];
  const float* bt = (const float*)d_in[5];
  const float* W2 = (const float*)d_in[6];
  const float* b2 = (const float*)d_in[7];
  float* out = (float*)d_out;

  const int nnz = in_sizes[1] / 2;
  const int* V = hei;
  const int* E = hei + nnz;

  char* ws = (char*)d_ws;
  size_t off = 0;
  auto alloc = [&](size_t bytes) -> void* {
    off = (off + 255) & ~(size_t)255;
    void* p = ws + off; off += bytes; return p;
  };
  float*  rsD = (float*)alloc((size_t)NN * 4);
  float*  iDE = (float*)alloc((size_t)MM * 4);
  float*  U   = (float*)alloc((size_t)NN * DF * 4);
  float*  G   = (float*)alloc((size_t)NN * DF * 4);
  float*  Xe  = (float*)alloc((size_t)MM * DF * 4);
  float*  Pe  = (float*)alloc((size_t)MM * DF * 4);
  float*  scl = (float*)alloc(8);                       // [0]=s, [1]=1/phi
  __bf16* Fbf = (__bf16*)alloc((size_t)NN * DF * 2);
  __bf16* W1s = (__bf16*)alloc((size_t)16 * 4 * 32 * 16 * 2);
  __bf16* W2s = (__bf16*)alloc((size_t)4 * 8 * 32 * 16 * 2);

  const int TB = 256;
  const int g_nnz = (nnz + TB - 1) / TB;
  const int g_nw  = (nnz + 7) / 8;               // 1 wave / incidence, 8 waves / block
  const int g_md  = (MM * DF + TB - 1) / TB;
  const int g_nd  = (NN * DF + TB - 1) / TB;

  // weight swizzle (once per launch)
  k_swz_w1<<<(16 * 4 * 32 * 16 + TB - 1) / TB, TB, 0, stream>>>(W1, W1s);
  k_swz_w2<<<(4 * 8 * 32 * 16 + TB - 1) / TB, TB, 0, stream>>>(W2, W2s);

  // degrees
  hipMemsetAsync(rsD, 0, (size_t)NN * 4, stream);
  hipMemsetAsync(iDE, 0, (size_t)MM * 4, stream);
  k_deg<<<g_nnz, TB, 0, stream>>>(V, E, rsD, iDE, nnz);
  k_invdeg<<<(NN + TB - 1) / TB, TB, 0, stream>>>(rsD, iDE);

  // initial edge sums + phi(X); U = X/phi, G holds un-normalized signal
  hipMemsetAsync(Xe, 0, (size_t)MM * DF * 4, stream);
  hipMemsetAsync(scl, 0, 8, stream);
  k_v2e<<<g_nw, TB, 0, stream>>>(x, V, E, rsD, Xe, nnz);
  k_reduce<<<g_md, TB, 0, stream>>>(Xe, iDE, scl);
  k_finphi<<<1, 1, 0, stream>>>(scl);
  k_initUG<<<g_nd, TB, 0, stream>>>(x, scl, U, G);

  for (int it = 0; it < 10; ++it) {
    k_pe<<<g_md, TB, 0, stream>>>(Xe, iDE, scl, Pe);          // Pe = V2E(F), F = G/phi
    hipMemsetAsync(G, 0, (size_t)NN * DF * 4, stream);
    k_e2v<<<g_nw, TB, 0, stream>>>(Pe, V, E, G, nnz);
    k_combine<<<g_nd, TB, 0, stream>>>(G, rsD, U);            // G = .9*E2V + .1*U
    hipMemsetAsync(Xe, 0, (size_t)MM * DF * 4, stream);
    hipMemsetAsync(scl, 0, 4, stream);                        // clear s, keep 1/phi
    k_v2e<<<g_nw, TB, 0, stream>>>(G, V, E, rsD, Xe, nnz);    // edge sums of new G
    k_reduce<<<g_md, TB, 0, stream>>>(Xe, iDE, scl);
    k_finphi<<<1, 1, 0, stream>>>(scl);                       // 1/phi(G)
  }

  // F = G/phi -> bf16, then fused WMMA decoder
  k_tobf<<<g_nd, TB, 0, stream>>>(G, scl, Fbf);
  k_decoder<<<NN / 16, 128, 0, stream>>>(Fbf, W1s, b1, gm, bt, W2s, b2, out);

  (void)n_in; (void)out_size; (void)ws_size;
}
